// VoiceCloningGenerator_55027120996532
// MI455X (gfx1250) — matrix operand order repaired and checked by reading, not compiled, not run on hardware
//
#include <hip/hip_runtime.h>

typedef __bf16 bf16_t;
typedef __attribute__((ext_vector_type(16))) __bf16 v16bf;
typedef __attribute__((ext_vector_type(8)))  __bf16 v8bf;
typedef __attribute__((ext_vector_type(4)))  __bf16 v4bf;
typedef __attribute__((ext_vector_type(8)))  float  v8f;

#define BATCH 64
#define LSEQ  512
#define TSEQ  1000
#define EDIM  256
#define HDIM  512
#define HDHALF 256
#define MELD  80
#define MELDP 128   // mel output dim padded to mult of 64 for GEMM
#define SDIM  128
#define KP0   608   // H+M=592 padded to mult of 32

// ---------------------------------------------------------------------------
// WMMA fragment helpers (layouts per CDNA5 ISA 7.12.2)
// A 16x32: lane<16 -> M=lane; elems 0..7 K=half*8+i, elems 8..15 K=16+half*8+i
// B 32x16 (as rows of (N,K) weight): lane%16 = N, elems same K order as A
// C/D 16x16: vgpr j, lane l -> M = j + 8*(l>>4), N = l&15
// ---------------------------------------------------------------------------
__device__ inline v16bf loadA_frag(const bf16_t* A, int lda, int m0, int k0, int lane) {
    int m = m0 + (lane & 15);
    int half = lane >> 4;
    const bf16_t* p = A + (size_t)m * lda + k0 + half * 8;
    v8bf lo = *(const v8bf*)p;
    v8bf hi = *(const v8bf*)(p + 16);
    return __builtin_shufflevector(lo, hi, 0,1,2,3,4,5,6,7,8,9,10,11,12,13,14,15);
}
__device__ inline v16bf loadB_frag(const bf16_t* W, int ldw, int n0, int k0, int lane) {
    int n = n0 + (lane & 15);
    int half = lane >> 4;
    const bf16_t* p = W + (size_t)n * ldw + k0 + half * 8;
    v8bf lo = *(const v8bf*)p;
    v8bf hi = *(const v8bf*)(p + 16);
    return __builtin_shufflevector(lo, hi, 0,1,2,3,4,5,6,7,8,9,10,11,12,13,14,15);
}
__device__ inline v8f wmma_bf16(v16bf a, v16bf b, v8f c) {
    return __builtin_amdgcn_wmma_f32_16x16x32_bf16(false, a, false, b, (short)0, c, false, false);
}
__device__ inline float sigm(float x) { return 1.f / (1.f + __expf(-x)); }

// ---------------------------------------------------------------------------
// WMMA GEMM: C[M,N] = act(A[M,K]_bf16 @ W[N,K]_bf16^T + bias)
// Wave tile = 32(M) x 64(N).  Requires M%32==0, N%64==0, K%32==0.
// Stores only columns n < Nstore (N may be zero-padded).
// outMode: 0=fp32 row-major, 1=bf16 row-major,
//          2=fp32 scattered (row m=t*64+b -> out[(b*T+t)*80+n])
// ---------------------------------------------------------------------------
__global__ void __launch_bounds__(256)
gemm_bf16_kernel(const bf16_t* __restrict__ A, const bf16_t* __restrict__ W,
                 const float* __restrict__ bias, void* __restrict__ Cout,
                 int M, int N, int K, int Nstore, int outMode, int relu)
{
    int lane = threadIdx.x & 31;
    int wave = threadIdx.x >> 5;
    int gw   = __builtin_amdgcn_readfirstlane(blockIdx.x * 8 + wave); // wave-uniform scalar
    int numMt = M >> 5;                 // 32-row tiles
    int numNg = N >> 6;                 // 64-col groups
    if (gw >= numMt * numNg) return;
    int mt = gw % numMt;
    int ng = gw / numMt;
    int m0 = mt << 5;
    int n0 = ng << 6;

    v8f acc[2][4] = {};
    for (int k0 = 0; k0 < K; k0 += 32) {
        v16bf a0 = loadA_frag(A, K, m0,      k0, lane);
        v16bf a1 = loadA_frag(A, K, m0 + 16, k0, lane);
#pragma unroll
        for (int j = 0; j < 4; ++j) {
            v16bf b = loadB_frag(W, K, n0 + j * 16, k0, lane);
            acc[0][j] = wmma_bf16(a0, b, acc[0][j]);
            acc[1][j] = wmma_bf16(a1, b, acc[1][j]);
        }
    }
    int half = lane >> 4, nl = lane & 15;
#pragma unroll
    for (int h = 0; h < 2; ++h) {
#pragma unroll
        for (int j = 0; j < 4; ++j) {
            int n = n0 + j * 16 + nl;
            if (n < Nstore) {
                float bv = bias ? bias[n] : 0.f;
#pragma unroll
                for (int r = 0; r < 8; ++r) {
                    int m = m0 + h * 16 + r + 8 * half;
                    float v = acc[h][j][r] + bv;
                    if (relu && v < 0.f) v = 0.f;
                    if (outMode == 0)      ((float*)Cout)[(size_t)m * Nstore + n] = v;
                    else if (outMode == 1) ((bf16_t*)Cout)[(size_t)m * Nstore + n] = (bf16_t)v;
                    else { int t = m >> 6, b = m & 63;
                           ((float*)Cout)[((size_t)b * TSEQ + t) * MELD + n] = v; }
                }
            }
        }
    }
}

// ---------------------------------------------------------------------------
// Persistent distributed LSTM scan.  NWG workgroups; WG wg owns hidden slice
// [wg*HS, wg*HS+HS).  Whh slice resident in LDS for the whole scan; per step
// the shared h (bf16) is staged into LDS, and the *next* step's Gx tile is
// double-buffer staged (non-temporal) so its latency hides behind the WMMA
// loop + pointwise update + software grid barrier.
// ---------------------------------------------------------------------------
template<int HID, int NWG>
__device__ void lstm_scan_dev(int wg,
    const bf16_t* __restrict__ Gx,    // (T, 64, 4*HID) bf16
    const bf16_t* __restrict__ Whh,   // (4*HID, HID)  bf16
    const float*  __restrict__ bias,  // (4*HID)
    bf16_t* __restrict__ hSh,         // (64, HID) bf16, zero-initialized
    float*  __restrict__ ysF,         // optional (T,64,HID) fp32
    bf16_t* __restrict__ ysB,         // optional (T,64,HID) bf16
    int T, int rev, unsigned* barCnt, unsigned* barGen, unsigned nwgBar)
{
    constexpr int HS = HID / NWG;       // 32
    constexpr int GC = 4 * HS;          // 128 gate columns per WG
    extern __shared__ char smem[];
    bf16_t* sWhh = (bf16_t*)smem;                        // GC x HID   bf16
    bf16_t* sH   = sWhh + (size_t)GC * HID;              // 64 x HID   bf16
    bf16_t* sGx0 = sH + (size_t)BATCH * HID;             // 64 x GC    bf16 (buf 0)
    bf16_t* sGx1 = sGx0 + (size_t)BATCH * GC;            // 64 x GC    bf16 (buf 1)
    float*  sG   = (float*)(sGx1 + (size_t)BATCH * GC);  // 64 x GC    fp32
    float*  sC   = sG + (size_t)BATCH * GC;              // 64 x HS    fp32

    int tid  = threadIdx.x;             // blockDim == 256 (8 waves)
    int lane = tid & 31;
    int wave = tid >> 5;

    auto stage_gx = [&](int row, bf16_t* dst) {
        const bf16_t* gxRow = Gx + (size_t)row * BATCH * (4 * HID);
        for (int idx = tid; idx < BATCH * GC / 4; idx += 256) {
            int m = idx / (GC / 4);
            int c4 = (idx % (GC / 4)) * 4;
            int g = c4 / HS, j = c4 % HS;
            v4bf v = __builtin_nontemporal_load(
                (const v4bf*)(gxRow + (size_t)m * (4 * HID) + g * HID + wg * HS + j));
            *(v4bf*)(dst + m * GC + c4) = v;
        }
    };

    // Cache Whh slice: local row r = g*HS + j <-> global row g*HID + wg*HS + j
    for (int idx = tid; idx < GC * (HID / 8); idx += 256) {
        int r = idx / (HID / 8), kk = (idx % (HID / 8)) * 8;
        int g = r / HS, j = r % HS;
        *(v8bf*)(sWhh + (size_t)r * HID + kk) =
            *(const v8bf*)(Whh + (size_t)(g * HID + wg * HS + j) * HID + kk);
    }
    for (int idx = tid; idx < BATCH * HS; idx += 256) sC[idx] = 0.f;
    for (int idx = tid; idx < BATCH * HID / 8; idx += 256)
        *(v8bf*)(sH + (size_t)idx * 8) = *(const v8bf*)(hSh + (size_t)idx * 8);
    stage_gx(rev ? (T - 1) : 0, sGx0);
    __syncthreads();

    // wave -> M-tile mt = wave/2, 4 consecutive N-tiles
    int mt = wave >> 1;
    int m0 = mt << 4;
    int ntBase = (wave & 1) * 4;
    int half = lane >> 4, nl = lane & 15;
    float bv[4];
#pragma unroll
    for (int j = 0; j < 4; ++j) {
        int c = (ntBase + j) * 16 + nl;
        bv[j] = bias[(c / HS) * HID + wg * HS + (c % HS)];
    }

    for (int s = 0; s < T; ++s) {
        int row = rev ? (T - 1 - s) : s;
        bf16_t* cur = (s & 1) ? sGx1 : sGx0;
        bf16_t* nxt = (s & 1) ? sGx0 : sGx1;

        v8f acc[4];
#pragma unroll
        for (int j = 0; j < 4; ++j) {
            int c = (ntBase + j) * 16 + nl;
#pragma unroll
            for (int r = 0; r < 8; ++r)
                acc[j][r] = bv[j] + (float)cur[(m0 + r + 8 * half) * GC + c];
        }
        if (s + 1 < T)                      // prefetch next step's Gx tile into LDS
            stage_gx(rev ? (T - 2 - s) : (s + 1), nxt);

        for (int k0 = 0; k0 < HID; k0 += 32) {
            v16bf a = loadA_frag(sH, HID, m0, k0, lane);
#pragma unroll
            for (int j = 0; j < 4; ++j) {
                v16bf b = loadB_frag(sWhh, HID, (ntBase + j) * 16, k0, lane);
                acc[j] = wmma_bf16(a, b, acc[j]);
            }
        }
#pragma unroll
        for (int j = 0; j < 4; ++j) {
            int c = (ntBase + j) * 16 + nl;
#pragma unroll
            for (int r = 0; r < 8; ++r)
                sG[(m0 + r + 8 * half) * GC + c] = acc[j][r];
        }
        __syncthreads();

        for (int idx = tid; idx < BATCH * HS; idx += 256) {
            int b = idx / HS, j = idx % HS;
            float gi = sG[b * GC + j];
            float gf = sG[b * GC + HS + j];
            float gg = sG[b * GC + 2 * HS + j];
            float go = sG[b * GC + 3 * HS + j];
            float cv = sigm(gf) * sC[idx] + sigm(gi) * tanhf(gg);
            sC[idx] = cv;
            float h = sigm(go) * tanhf(cv);
            int hcol = wg * HS + j;
            hSh[(size_t)b * HID + hcol] = (bf16_t)h;
            if (ysF) ysF[((size_t)row * BATCH + b) * HID + hcol] = h;
            if (ysB) ysB[((size_t)row * BATCH + b) * HID + hcol] = (bf16_t)h;
        }
        __threadfence();
        __syncthreads();
        if (tid == 0) {                 // software grid barrier (NWG resident WGs)
            volatile unsigned* vGen = (volatile unsigned*)barGen;
            unsigned gen = *vGen;
            unsigned arr = atomicAdd(barCnt, 1u);
            if (arr == nwgBar - 1u) { *barCnt = 0u; __threadfence(); atomicAdd(barGen, 1u); }
            else { while (*vGen == gen) __builtin_amdgcn_s_sleep(1); }
        }
        __syncthreads();
        __threadfence();
        if (s + 1 < T) {   // restage shared h for next step
            for (int idx = tid; idx < BATCH * HID / 8; idx += 256)
                *(v8bf*)(sH + (size_t)idx * 8) = *(const v8bf*)(hSh + (size_t)idx * 8);
        }
        __syncthreads();
    }
}

#define ENC_LDS (size_t)(128*HDHALF*2 + BATCH*HDHALF*2 + 2*BATCH*128*2 + BATCH*128*4 + BATCH*32*4)
#define DEC_LDS (size_t)(128*HDIM*2   + BATCH*HDIM*2   + 2*BATCH*128*2 + BATCH*128*4 + BATCH*32*4)

__global__ void __launch_bounds__(256)
enc_scan_kernel(const bf16_t* GxF, const bf16_t* GxB,
                const bf16_t* WhhF, const bf16_t* WhhB,
                const float* bF, const float* bB,
                bf16_t* hShF, bf16_t* hShB, float* ysFwd, float* ysBwd,
                unsigned* bars)
{
    if (blockIdx.x < 8)
        lstm_scan_dev<HDHALF, 8>(blockIdx.x, GxF, WhhF, bF, hShF, ysFwd, nullptr,
                                 LSEQ, 0, bars + 0, bars + 1, 8u);
    else
        lstm_scan_dev<HDHALF, 8>(blockIdx.x - 8, GxB, WhhB, bB, hShB, ysBwd, nullptr,
                                 LSEQ, 1, bars + 2, bars + 3, 8u);
}

__global__ void __launch_bounds__(256)
dec_scan_kernel(const bf16_t* Gx, const bf16_t* Whh, const float* bias,
                bf16_t* hSh, bf16_t* ysB, unsigned* barCnt, unsigned* barGen)
{
    lstm_scan_dev<HDIM, 16>(blockIdx.x, Gx, Whh, bias, hSh, nullptr, ysB,
                            TSEQ, 0, barCnt, barGen, 16u);
}

// ---------------------------------------------------------------------------
// Small elementwise / gather kernels
// ---------------------------------------------------------------------------
__global__ void f32_to_bf16_kernel(const float* __restrict__ s, bf16_t* __restrict__ d, long n) {
    long i = (long)blockIdx.x * blockDim.x + threadIdx.x;
    if (i < n) d[i] = (bf16_t)s[i];
}
// pad K columns (dec_Wih0: 592 -> 608)
__global__ void f32_to_bf16_padk_kernel(const float* __restrict__ s, bf16_t* __restrict__ d,
                                        int rows, int srcK, int dstK) {
    long i = (long)blockIdx.x * blockDim.x + threadIdx.x;
    if (i >= (long)rows * dstK) return;
    int r = (int)(i / dstK), k = (int)(i % dstK);
    d[i] = (k < srcK) ? (bf16_t)s[(size_t)r * srcK + k] : (bf16_t)0.f;
}
// pad rows (mel_W2: 80 -> 128 rows of K=512)
__global__ void f32_to_bf16_padr_kernel(const float* __restrict__ s, bf16_t* __restrict__ d,
                                        int srcRows, int dstRows, int K) {
    long i = (long)blockIdx.x * blockDim.x + threadIdx.x;
    if (i >= (long)dstRows * K) return;
    int r = (int)(i / K);
    d[i] = (r < srcRows) ? (bf16_t)s[i] : (bf16_t)0.f;
}
__global__ void embed_gather_kernel(const int* __restrict__ text, const float* __restrict__ embed,
                                    bf16_t* __restrict__ X) {
    long i = (long)blockIdx.x * blockDim.x + threadIdx.x;   // (L*B) x E
    if (i >= (long)LSEQ * BATCH * EDIM) return;
    int e = (int)(i & (EDIM - 1));
    int tb = (int)(i >> 8);
    int b = tb & (BATCH - 1);
    int t = tb >> 6;
    int tok = text[b * LSEQ + t];
    X[i] = (bf16_t)embed[(size_t)tok * EDIM + e];
}
// rnn_in[(t*64+b), 0:80]=mel_in, [80:336]=ysF+sp, [336:592]=ysB+sp, [592:608]=0
__global__ void build_rnn_in_kernel(const float* __restrict__ mel, const float* __restrict__ ysF,
                                    const float* __restrict__ ysB, const float* __restrict__ sp,
                                    bf16_t* __restrict__ out) {
    long i = (long)blockIdx.x * blockDim.x + threadIdx.x;
    if (i >= (long)TSEQ * BATCH * KP0) return;
    int c = (int)(i % KP0);
    int tb = (int)(i / KP0);
    int b = tb & (BATCH - 1);
    int t = tb >> 6;
    float v;
    if (c < MELD)      v = (t == 0) ? 0.f : mel[((size_t)b * TSEQ + (t - 1)) * MELD + c];
    else if (c < 336)  { int tl = t & (LSEQ - 1);
                         v = ysF[((size_t)tl * BATCH + b) * HDHALF + (c - 80)] + sp[b * HDIM + (c - 80)]; }
    else if (c < 592)  { int tl = t & (LSEQ - 1);
                         v = ysB[((size_t)tl * BATCH + b) * HDHALF + (c - 336)] + sp[b * HDIM + 256 + (c - 336)]; }
    else v = 0.f;
    out[i] = (bf16_t)v;
}
__global__ void stop_kernel(const bf16_t* __restrict__ h2, const float* __restrict__ sw,
                            const float* __restrict__ sb, float* __restrict__ out) {
    int row = blockIdx.x * 8 + (threadIdx.x >> 5);   // rows = t*64+b
    int lane = threadIdx.x & 31;
    if (row >= TSEQ * BATCH) return;
    float a = 0.f;
    for (int k = lane; k < HDIM; k += 32) a += (float)h2[(size_t)row * HDIM + k] * sw[k];
#pragma unroll
    for (int off = 16; off > 0; off >>= 1) a += __shfl_down(a, off);
    if (lane == 0) {
        int b = row & 63, t = row >> 6;
        out[(size_t)BATCH * TSEQ * MELD + (size_t)b * TSEQ + t] = a + sb[0];
    }
}

// ---------------------------------------------------------------------------
// Host launcher
// ---------------------------------------------------------------------------
static inline void launch_gemm(const bf16_t* A, const bf16_t* W, const float* bias, void* C,
                               int M, int N, int K, int Nstore, int outMode, int relu,
                               hipStream_t s) {
    long waves = (long)(M >> 5) * (N >> 6);
    int blocks = (int)((waves + 7) / 8);
    gemm_bf16_kernel<<<blocks, 256, 0, s>>>(A, W, bias, C, M, N, K, Nstore, outMode, relu);
}

extern "C" void kernel_launch(void* const* d_in, const int* in_sizes, int n_in,
                              void* d_out, int out_size, void* d_ws, size_t ws_size,
                              hipStream_t stream) {
    const int*   text   = (const int*)  d_in[0];
    const float* spk    = (const float*)d_in[1];
    const float* tmel   = (const float*)d_in[2];
    const float* embed  = (const float*)d_in[3];
    const float* eWihF  = (const float*)d_in[4];
    const float* eWhhF  = (const float*)d_in[5];
    const float* ebF    = (const float*)d_in[6];
    const float* eWihB  = (const float*)d_in[7];
    const float* eWhhB  = (const float*)d_in[8];
    const float* ebB    = (const float*)d_in[9];
    const float* spW1   = (const float*)d_in[10];
    const float* spb1   = (const float*)d_in[11];
    const float* spW2   = (const float*)d_in[12];
    const float* spb2   = (const float*)d_in[13];
    const float* dWih0  = (const float*)d_in[14];
    const float* dWhh0  = (const float*)d_in[15];
    const float* db0    = (const float*)d_in[16];
    const float* dWih1  = (const float*)d_in[17];
    const float* dWhh1  = (const float*)d_in[18];
    const float* db1    = (const float*)d_in[19];
    const float* mW1    = (const float*)d_in[20];
    const float* mb1    = (const float*)d_in[21];
    const float* mW2    = (const float*)d_in[22];
    const float* mb2    = (const float*)d_in[23];
    const float* stW    = (const float*)d_in[24];
    const float* stB    = (const float*)d_in[25];

    char* base = (char*)d_ws;
    size_t off = 0;
    auto alloc = [&](size_t bytes) -> char* {
        char* p = base + off;
        off = (off + bytes + 255) & ~(size_t)255;
        return p;
    };
    // --- zero region (memset each launch): barriers + shared h states ---
    unsigned* bars   = (unsigned*)alloc(32);                 // 8 counters
    bf16_t* hShF     = (bf16_t*)alloc((size_t)BATCH * HDHALF * 2);
    bf16_t* hShB     = (bf16_t*)alloc((size_t)BATCH * HDHALF * 2);
    bf16_t* hShD0    = (bf16_t*)alloc((size_t)BATCH * HDIM * 2);
    bf16_t* hShD1    = (bf16_t*)alloc((size_t)BATCH * HDIM * 2);
    size_t zbytes = off;
    // --- bf16 weights ---
    bf16_t* wEWihF = (bf16_t*)alloc((size_t)1024 * 256 * 2);
    bf16_t* wEWhhF = (bf16_t*)alloc((size_t)1024 * 256 * 2);
    bf16_t* wEWihB = (bf16_t*)alloc((size_t)1024 * 256 * 2);
    bf16_t* wEWhhB = (bf16_t*)alloc((size_t)1024 * 256 * 2);
    bf16_t* wSpW1  = (bf16_t*)alloc((size_t)512 * 128 * 2);
    bf16_t* wSpW2  = (bf16_t*)alloc((size_t)512 * 512 * 2);
    bf16_t* wDWih0 = (bf16_t*)alloc((size_t)2048 * KP0 * 2);
    bf16_t* wDWhh0 = (bf16_t*)alloc((size_t)2048 * 512 * 2);
    bf16_t* wDWih1 = (bf16_t*)alloc((size_t)2048 * 512 * 2);
    bf16_t* wDWhh1 = (bf16_t*)alloc((size_t)2048 * 512 * 2);
    bf16_t* wMW1   = (bf16_t*)alloc((size_t)512 * 512 * 2);
    bf16_t* wMW2   = (bf16_t*)alloc((size_t)MELDP * 512 * 2);   // row-padded
    bf16_t* spkBf  = (bf16_t*)alloc((size_t)BATCH * SDIM * 2);
    // --- activations ---
    bf16_t* Xemb  = (bf16_t*)alloc((size_t)LSEQ * BATCH * EDIM * 2);
    bf16_t* GxF   = (bf16_t*)alloc((size_t)LSEQ * BATCH * 1024 * 2);
    bf16_t* GxB   = (bf16_t*)alloc((size_t)LSEQ * BATCH * 1024 * 2);
    float*  ysF   = (float*) alloc((size_t)LSEQ * BATCH * HDHALF * 4);
    float*  ysB   = (float*) alloc((size_t)LSEQ * BATCH * HDHALF * 4);
    bf16_t* sp1   = (bf16_t*)alloc((size_t)BATCH * HDIM * 2);
    float*  spF   = (float*) alloc((size_t)BATCH * HDIM * 4);
    bf16_t* rnnIn = (bf16_t*)alloc((size_t)TSEQ * BATCH * KP0 * 2);
    bf16_t* Gx01  = (bf16_t*)alloc((size_t)TSEQ * BATCH * 2048 * 2);  // reused L0/L1
    bf16_t* h1seq = (bf16_t*)alloc((size_t)TSEQ * BATCH * HDIM * 2);
    bf16_t* h2seq = (bf16_t*)alloc((size_t)TSEQ * BATCH * HDIM * 2);
    bf16_t* mh    = (bf16_t*)alloc((size_t)TSEQ * BATCH * HDIM * 2);

    hipMemsetAsync(d_ws, 0, zbytes, stream);

    auto conv = [&](const float* s, bf16_t* d, long n) {
        f32_to_bf16_kernel<<<(unsigned)((n + 255) / 256), 256, 0, stream>>>(s, d, n);
    };
    conv(eWihF, wEWihF, 1024L * 256);  conv(eWhhF, wEWhhF, 1024L * 256);
    conv(eWihB, wEWihB, 1024L * 256);  conv(eWhhB, wEWhhB, 1024L * 256);
    conv(spW1,  wSpW1,  512L * 128);   conv(spW2,  wSpW2,  512L * 512);
    conv(dWhh0, wDWhh0, 2048L * 512);  conv(dWih1, wDWih1, 2048L * 512);
    conv(dWhh1, wDWhh1, 2048L * 512);  conv(mW1,   wMW1,   512L * 512);
    conv(spk,   spkBf,  (long)BATCH * SDIM);
    {
        long n = 2048L * KP0;
        f32_to_bf16_padk_kernel<<<(unsigned)((n + 255) / 256), 256, 0, stream>>>(
            dWih0, wDWih0, 2048, 592, KP0);
    }
    {
        long n = (long)MELDP * 512;
        f32_to_bf16_padr_kernel<<<(unsigned)((n + 255) / 256), 256, 0, stream>>>(
            mW2, wMW2, MELD, MELDP, 512);
    }
    {
        long n = (long)LSEQ * BATCH * EDIM;
        embed_gather_kernel<<<(unsigned)((n + 255) / 256), 256, 0, stream>>>(text, embed, Xemb);
    }
    // Encoder input projections (time-parallel GEMMs)
    launch_gemm(Xemb, wEWihF, nullptr, GxF, LSEQ * BATCH, 1024, EDIM, 1024, 1, 0, stream);
    launch_gemm(Xemb, wEWihB, nullptr, GxB, LSEQ * BATCH, 1024, EDIM, 1024, 1, 0, stream);
    // Bidirectional encoder scan: 16 persistent WGs (8 per direction)
    enc_scan_kernel<<<16, 256, ENC_LDS, stream>>>(GxF, GxB, wEWhhF, wEWhhB, ebF, ebB,
                                                  hShF, hShB, ysF, ysB, bars);
    // Speaker MLP
    launch_gemm(spkBf, wSpW1, spb1, sp1, BATCH, HDIM, SDIM, HDIM, 1, 1, stream);
    launch_gemm(sp1,   wSpW2, spb2, spF, BATCH, HDIM, HDIM, HDIM, 0, 0, stream);
    // Decoder input assembly + layer-0 input projection
    {
        long n = (long)TSEQ * BATCH * KP0;
        build_rnn_in_kernel<<<(unsigned)((n + 255) / 256), 256, 0, stream>>>(tmel, ysF, ysB, spF, rnnIn);
    }
    launch_gemm(rnnIn, wDWih0, nullptr, Gx01, TSEQ * BATCH, 2048, KP0, 2048, 1, 0, stream);
    dec_scan_kernel<<<16, 256, DEC_LDS, stream>>>(Gx01, wDWhh0, db0, hShD0, h1seq, bars + 4, bars + 5);
    launch_gemm(h1seq, wDWih1, nullptr, Gx01, TSEQ * BATCH, 2048, HDIM, 2048, 1, 0, stream);
    dec_scan_kernel<<<16, 256, DEC_LDS, stream>>>(Gx01, wDWhh1, db1, hShD1, h2seq, bars + 6, bars + 7);
    // Output heads
    launch_gemm(h2seq, wMW1, mb1, mh, TSEQ * BATCH, HDIM, HDIM, HDIM, 1, 1, stream);
    launch_gemm(mh, wMW2, mb2, d_out, TSEQ * BATCH, MELDP, HDIM, MELD, 2, 0, stream);
    stop_kernel<<<(TSEQ * BATCH) / 8, 256, 0, stream>>>(h2seq, stW, stB, (float*)d_out);
    (void)in_sizes; (void)n_in; (void)out_size; (void)ws_size;
}
